// RNN_33354716021342
// MI455X (gfx1250) — compile-verified, matrix operand order checked by reading, not run on hardware
//
#include <hip/hip_runtime.h>

#define NHID 512
#define NTOK 32000
#define TT   64
#define BB   32
#define GBLK 16          // persistent blocks in the scan (16 WGPs share L2 bandwidth)
#define SN   (BB * NHID) // 16384 elements per state

typedef __attribute__((ext_vector_type(16))) __bf16 v16bf;
typedef __attribute__((ext_vector_type(8)))  __bf16 v8bf;
typedef __attribute__((ext_vector_type(8)))  float  v8f;

__device__ inline __bf16 f2bf(float f) {
  unsigned u = __builtin_bit_cast(unsigned, f);
  unsigned r = (u + 0x7FFFu + ((u >> 16) & 1u)) >> 16;
  unsigned short s = (unsigned short)r;
  return __builtin_bit_cast(__bf16, s);
}
__device__ inline float bf2f(__bf16 b) {
  unsigned short s = __builtin_bit_cast(unsigned short, b);
  unsigned u = ((unsigned)s) << 16;
  return __builtin_bit_cast(float, u);
}
__device__ inline float sigm(float x) { return 1.f / (1.f + expf(-x)); }

// bf16 16x16x32 fragment load: row-major source, row = M (A) or N (pre-transposed B),
// contiguous K. Lane<16: K in {0..7,16..23}; lane>=16: K in {8..15,24..31} (ISA 7.12.2).
__device__ inline v16bf ldfrag(const __bf16* p0, int stride, int lane) {
  int half = lane >> 4, r = lane & 15;
  const __bf16* p = p0 + (size_t)r * stride;
  v8bf lo = *(const v8bf*)(p + half * 8);
  v8bf hi = *(const v8bf*)(p + 16 + half * 8);
  return __builtin_shufflevector(lo, hi, 0,1,2,3,4,5,6,7,8,9,10,11,12,13,14,15);
}
__device__ inline v8f wmma_bf(v16bf a, v16bf b, v8f c) {
  return __builtin_amdgcn_wmma_f32_16x16x32_bf16(false, a, false, b, (short)0, c,
                                                 false, false);
}

// ---- CDNA5 async global->LDS copy (ASYNCcnt-tracked, ISA 15.18.3 op 98) ----
// VDST operand = LDS byte address; per ISA aperture translation the low 32 bits of a
// generic pointer to a __shared__ object are exactly that LDS address.
__device__ inline void async_ld_b128(unsigned lds_off, const void* gaddr) {
  asm volatile("global_load_async_to_lds_b128 %0, %1, off"
               :: "v"(lds_off), "v"(gaddr)
               : "memory");
}
__device__ inline void wait_async_le12() {
  asm volatile("s_wait_asynccnt 12" ::: "memory");
}
__device__ inline void wait_async_0() {
  asm volatile("s_wait_asynccnt 0" ::: "memory");
}

// agent-scope barrier for the 16 persistent scan blocks
__device__ inline void gbar(unsigned* cnt, unsigned target) {
  __threadfence();
  __syncthreads();
  if (threadIdx.x == 0) {
    __hip_atomic_fetch_add(cnt, 1u, __ATOMIC_ACQ_REL, __HIP_MEMORY_SCOPE_AGENT);
    while (__hip_atomic_load(cnt, __ATOMIC_ACQUIRE, __HIP_MEMORY_SCOPE_AGENT) < target)
      __builtin_amdgcn_s_sleep(2);
  }
  __syncthreads();
  __threadfence();   // invalidate L0 so fresh cross-WGP state is observed
}

// ---------------- conversion / gather kernels ----------------
__global__ void k_w0t(const float* __restrict__ W0, __bf16* __restrict__ out) {
  int idx = blockIdx.x * 256 + threadIdx.x;          // idx = n*1024 + k
  int n = idx >> 10, k = idx & 1023;
  out[idx] = f2bf(W0[(size_t)k * 1024 + n]);         // W0T[n][k] = W0[k][n]
}
__global__ void k_wst(const float* __restrict__ Ws, __bf16* __restrict__ out) {
  size_t idx = (size_t)blockIdx.x * 256 + threadIdx.x; // ((ij*1024)+n)*512 + k
  int k  = (int)(idx & 511);
  int n  = (int)((idx >> 9) & 1023);
  int ij = (int)(idx >> 19);
  out[idx] = f2bf(Ws[((size_t)ij * 512 + k) * 1024 + n]);
}
__global__ void k_tab(const float* __restrict__ Tf, __bf16* __restrict__ out) {
  size_t idx = (size_t)blockIdx.x * 256 + threadIdx.x;
  out[idx] = f2bf(Tf[idx]);
}
__global__ void k_embed(const int* __restrict__ ids, const float* __restrict__ Tf,
                        __bf16* __restrict__ out) {
  int idx = blockIdx.x * 256 + threadIdx.x;          // r*512 + k
  int r = idx >> 9, k = idx & 511;
  out[idx] = f2bf(Tf[(size_t)ids[r] * NHID + k]);
}
__global__ void k_lasth(const float* __restrict__ ghid, float* __restrict__ out) {
  int idx = blockIdx.x * 256 + threadIdx.x;
  out[idx] = ghid[(size_t)(TT - 1) * SN + idx];
}

// ---------------- persistent RNN scan (16 blocks, 1024 threads) ----------------
__global__ __launch_bounds__(1024) void rnn_scan(
    const float*  __restrict__ h0,
    const __bf16* __restrict__ emb,    // [T][32][512] bf16
    const __bf16* __restrict__ W0T,    // [1024 (n)][1024 (k)]
    const __bf16* __restrict__ WsT,    // [64 (i*8+j)][1024 (n)][512 (k)]
    float*        __restrict__ gs,     // [8][32*512] state accumulators
    float*        __restrict__ ghid,   // [64][32*512]
    __bf16*       __restrict__ ghbf,   // [64][32*512]
    unsigned*     __restrict__ cnt) {
  __shared__ __attribute__((aligned(16))) __bf16 A[BB * 1024];
  const int tid  = threadIdx.x;
  const int lane = tid & 31;
  const int w    = tid >> 5;
  const int bx   = blockIdx.x;         // owns output column tiles nc = {2bx, 2bx+1}
  unsigned phase = 0;

  for (int e = 0; e < 16; ++e) {       // initial hidden -> A[:,512:]
    int idx = tid * 16 + e;
    int b = idx >> 9, n = idx & 511;
    A[b * 1024 + 512 + n] = f2bf(h0[idx]);
  }

  for (int t = 0; t < TT; ++t) {
    const __bf16* xt = emb + (size_t)t * SN;
    for (int e = 0; e < 16; ++e) {     // x_t -> A[:,:512]
      int idx = tid * 16 + e;
      int b = idx >> 9, n = idx & 511;
      A[b * 1024 + n] = xt[idx];
    }
    __syncthreads();

    // ---- node 0: [xh] @ W0, s0 = h + sig(c0)*(tanh(h0)-h) ----
    if (w < 4) {                       // wave-uniform -> EXEC all ones for WMMA
      int m = w >> 1, ncl = w & 1;
      int nc = bx * 2 + ncl;
      const __bf16* bc = W0T + (size_t)(nc * 16) * 1024;
      const __bf16* bh = W0T + (size_t)(512 + nc * 16) * 1024;
      const __bf16* ap = &A[(m * 16) * 1024];
      v8f cc = {}, ch = {};
      for (int kb = 0; kb < 32; ++kb) {
        v16bf a  = ldfrag(ap + kb * 32, 1024, lane);
        v16bf b0 = ldfrag(bc + kb * 32, 1024, lane);
        v16bf b1 = ldfrag(bh + kb * 32, 1024, lane);
        cc = wmma_bf(a, b0, cc);
        ch = wmma_bf(a, b1, ch);
      }
      int N  = nc * 16 + (lane & 15);
      int rb = m * 16 + ((lane >> 4) << 3);
#pragma unroll
      for (int r = 0; r < 8; ++r) {
        int brow = rb + r;
        float hp = bf2f(A[brow * 1024 + 512 + N]);
        gs[brow * NHID + N] = hp + sigm(cc[r]) * (tanhf(ch[r]) - hp);
      }
    }
    ++phase; gbar(cnt, phase * GBLK);

    // ---- DAG rounds i = 0..6: edges (i, j>i) ----
    for (int i = 0; i < 7; ++i) {
      float inv = (i == 0) ? 1.f : 1.f / (float)i;   // counts[i] = max(i,1)
      const float* gsi = gs + (size_t)i * SN;
      for (int e = 0; e < 16; ++e) {   // full s_prev -> A[:,:512] bf16
        int idx = tid * 16 + e;
        int b = idx >> 9, n = idx & 511;
        A[b * 1024 + n] = f2bf(gsi[idx] * inv);
      }
      __syncthreads();

      int npairs = (7 - i) * 4;        // (#edges) x (2 M-tiles x 2 owned N-tiles)
      if (w < npairs) {                // wave-uniform
        int j   = i + 1 + (w >> 2);
        int q   = w & 3;
        int m   = q >> 1, ncl = q & 1;
        int nc  = bx * 2 + ncl;
        const __bf16* Wb = WsT + ((size_t)(i * 8 + j) << 19);
        const __bf16* bc = Wb + (size_t)(nc * 16) * 512;
        const __bf16* bh = Wb + (size_t)(512 + nc * 16) * 512;
        const __bf16* ap = &A[(m * 16) * 1024];
        v8f cc = {}, ch = {};
        for (int kb = 0; kb < 16; ++kb) {
          v16bf a  = ldfrag(ap + kb * 32, 1024, lane);
          v16bf b0 = ldfrag(bc + kb * 32, 512, lane);
          v16bf b1 = ldfrag(bh + kb * 32, 512, lane);
          cc = wmma_bf(a, b0, cc);
          ch = wmma_bf(a, b1, ch);
        }
        int N  = nc * 16 + (lane & 15);
        int rb = m * 16 + ((lane >> 4) << 3);
        float* gsj = gs + (size_t)j * SN;
#pragma unroll
        for (int r = 0; r < 8; ++r) {
          int brow = rb + r;
          float sp = bf2f(A[brow * 1024 + N]);
          float hh = ch[r];
          float av;
          if (j == 7) av = hh;                      // identity
          else {
            int mm = (j - 1) % 3;                   // tanh, relu, sigmoid pattern
            av = (mm == 0) ? tanhf(hh) : (mm == 1) ? fmaxf(hh, 0.f) : sigm(hh);
          }
          float s  = sp + sigm(cc[r]) * (av - sp);
          int off  = brow * NHID + N;
          if (i == 0) gsj[off] = s; else gsj[off] += s;  // unique owner per element
        }
      }
      ++phase; gbar(cnt, phase * GBLK);
    }

    // ---- h_new = mean_j (states[j]/j); full copy into LDS, owned slice to global ----
    int clo = bx * 32, chi = clo + 32;
    for (int e = 0; e < 16; ++e) {
      int idx = tid * 16 + e;
      int b = idx >> 9, n = idx & 511;
      float acc = 0.f;
#pragma unroll
      for (int j = 1; j <= 7; ++j) acc += gs[(size_t)j * SN + idx] * (1.f / (float)j);
      float hn = acc * (1.f / 7.f);
      A[b * 1024 + 512 + n] = f2bf(hn);
      if (n >= clo && n < chi) {
        ghid[(size_t)t * SN + idx] = hn;
        ghbf[(size_t)t * SN + idx] = f2bf(hn);
      }
    }
    __syncthreads();
  }
}

// ---------------- decoder: logits = H @ table^T + bias ----------------
// 64(M) x 128(N) per workgroup; K=512 in 4 chunks of 128, async-staged into LDS
// (global_load_async_to_lds_b128) with double buffering and s_wait_asynccnt pipelining.
__global__ __launch_bounds__(256) void k_decoder(
    const __bf16* __restrict__ H,      // [2048][512]
    const __bf16* __restrict__ Tb,     // [32000][512] (row = vocab = output col)
    const float*  __restrict__ bias,
    float*        __restrict__ out) {
  __shared__ __attribute__((aligned(16))) __bf16 Ash[2][64 * 128];
  __shared__ __attribute__((aligned(16))) __bf16 Bsh[2][128 * 128];
  int tid = threadIdx.x, lane = tid & 31, w = tid >> 5;
  int R0 = blockIdx.y * 64;
  int C0 = blockIdx.x * 128;
  const __bf16* Arow = H  + (size_t)R0 * NHID;
  const __bf16* Brow = Tb + (size_t)C0 * NHID;

  auto issue = [&](int buf, int kc) {
    // A chunk: 64 rows x 128 k = 1024 x 16B; 4 issues/thread
#pragma unroll
    for (int rr = 0; rr < 4; ++rr) {
      int lin = tid + rr * 256;
      int row = lin >> 4, c = lin & 15;
      unsigned loff = (unsigned)(uintptr_t)&Ash[buf][row * 128 + c * 8];
      async_ld_b128(loff, Arow + (size_t)row * NHID + kc * 128 + c * 8);
    }
    // B chunk: 128 rows x 128 k = 2048 x 16B; 8 issues/thread
#pragma unroll
    for (int rr = 0; rr < 8; ++rr) {
      int lin = tid + rr * 256;
      int row = lin >> 4, c = lin & 15;
      unsigned loff = (unsigned)(uintptr_t)&Bsh[buf][row * 128 + c * 8];
      async_ld_b128(loff, Brow + (size_t)row * NHID + kc * 128 + c * 8);
    }
  };

  v8f acc[4] = {};
  issue(0, 0);
  for (int kc = 0; kc < 4; ++kc) {
    int buf = kc & 1;
    if (kc < 3) { issue(buf ^ 1, kc + 1); wait_async_le12(); }  // 12 newer outstanding
    else        { wait_async_0(); }
    __syncthreads();                    // all waves' chunk data in LDS
    const __bf16* Ab = &Ash[buf][(w & 3) * 16 * 128];
    const __bf16* Bb = &Bsh[buf][(w >> 2) * 64 * 128];
    for (int kb = 0; kb < 4; ++kb) {
      v16bf a = ldfrag(Ab + kb * 32, 128, lane);
#pragma unroll
      for (int tt = 0; tt < 4; ++tt) {
        v16bf b = ldfrag(Bb + (tt * 16) * 128 + kb * 32, 128, lane);
        acc[tt] = wmma_bf(a, b, acc[tt]);
      }
    }
    __syncthreads();                    // buffer reusable for kc+2's async writes
  }

#pragma unroll
  for (int tt = 0; tt < 4; ++tt) {
    int col = C0 + (w >> 2) * 64 + tt * 16 + (lane & 15);
    float bs = bias[col];
#pragma unroll
    for (int r = 0; r < 8; ++r) {
      int row = R0 + (w & 3) * 16 + r + ((lane >> 4) << 3);
      out[(size_t)row * NTOK + col] = acc[tt][r] + bs;
    }
  }
}

// ---------------- in-place log-softmax over 32000 per row ----------------
__global__ __launch_bounds__(1024) void k_logsoftmax(float* __restrict__ out) {
  __shared__ float red[32];
  float* p = out + (size_t)blockIdx.x * NTOK;
  int tid = threadIdx.x;
  float mx = -INFINITY;
  for (int v = tid; v < NTOK; v += 1024) mx = fmaxf(mx, p[v]);
  for (int o = 16; o > 0; o >>= 1) mx = fmaxf(mx, __shfl_xor(mx, o, 32));
  if (!(tid & 31)) red[tid >> 5] = mx;
  __syncthreads();
  if (tid < 32) {
    float v = red[tid];
    for (int o = 16; o > 0; o >>= 1) v = fmaxf(v, __shfl_xor(v, o, 32));
    red[tid] = v;
  }
  __syncthreads();
  mx = red[0];
  float sm = 0.f;
  for (int v = tid; v < NTOK; v += 1024) sm += expf(p[v] - mx);
  for (int o = 16; o > 0; o >>= 1) sm += __shfl_xor(sm, o, 32);
  __syncthreads();
  if (!(tid & 31)) red[tid >> 5] = sm;
  __syncthreads();
  if (tid < 32) {
    float v = red[tid];
    for (int o = 16; o > 0; o >>= 1) v += __shfl_xor(v, o, 32);
    red[tid] = v;
  }
  __syncthreads();
  float lse = mx + logf(red[0]);
  for (int v = tid; v < NTOK; v += 1024) p[v] -= lse;
}

extern "C" void kernel_launch(void* const* d_in, const int* in_sizes, int n_in,
                              void* d_out, int out_size, void* d_ws, size_t ws_size,
                              hipStream_t stream) {
  (void)in_sizes; (void)n_in; (void)out_size; (void)ws_size;
  const int*   ids    = (const int*)d_in[0];
  const float* hidden = (const float*)d_in[1];
  const float* table  = (const float*)d_in[2];
  const float* W0     = (const float*)d_in[3];
  const float* Ws     = (const float*)d_in[4];
  const float* bias   = (const float*)d_in[5];
  float* out = (float*)d_out;

  char* ws = (char*)d_ws;
  size_t off = 0;
  auto take = [&](size_t bytes) { char* p = ws + off; off = (off + bytes + 255) & ~(size_t)255; return p; };
  __bf16*   w0t  = (__bf16*)take((size_t)1024 * 1024 * 2);
  __bf16*   wst  = (__bf16*)take((size_t)64 * 1024 * 512 * 2);
  __bf16*   tb   = (__bf16*)take((size_t)NTOK * NHID * 2);
  __bf16*   embb = (__bf16*)take((size_t)TT * SN * 2);
  float*    gs   = (float*)take((size_t)8 * SN * 4);
  float*    ghid = (float*)take((size_t)TT * SN * 4);
  __bf16*   ghbf = (__bf16*)take((size_t)TT * SN * 2);
  unsigned* cnt  = (unsigned*)take(256);

  hipMemsetAsync(cnt, 0, sizeof(unsigned), stream);
  k_w0t  <<<(1024 * 1024) / 256, 256, 0, stream>>>(W0, w0t);
  k_wst  <<<(64 * 1024 * 512) / 256, 256, 0, stream>>>(Ws, wst);
  k_tab  <<<(NTOK * NHID) / 256, 256, 0, stream>>>(table, tb);
  k_embed<<<(TT * SN) / 256, 256, 0, stream>>>(ids, table, embb);

  rnn_scan<<<GBLK, 1024, 0, stream>>>(hidden, embb, w0t, wst, gs, ghid, ghbf, cnt);

  k_decoder<<<dim3(NTOK / 128, (TT * BB) / 64), 256, 0, stream>>>(ghbf, tb, bias, out);
  k_logsoftmax<<<TT * BB, 1024, 0, stream>>>(out);
  k_lasth<<<SN / 256, 256, 0, stream>>>(ghid, out + (size_t)TT * BB * NTOK);
}
